// MatPhys_GA_41540923687403
// MI455X (gfx1250) — matrix operand order repaired and checked by reading, not compiled, not run on hardware
//
#include <hip/hip_runtime.h>
#include <math.h>

#define NTOK 60
#define NP   64
#define DD   256
#define HH   4
#define DHD  64
#define DFF  1024

#if defined(__has_builtin)
#  if __has_builtin(__builtin_amdgcn_global_load_async_to_lds_b128)
#    define ASYNC_LDS 1
#  endif
#  if __has_builtin(__builtin_amdgcn_s_wait_asynccnt)
#    define HAVE_WAIT_ASYNC 1
#  endif
#endif

typedef _Float16 v16h __attribute__((ext_vector_type(16)));
typedef _Float16 v8h  __attribute__((ext_vector_type(8)));
typedef float    v8f  __attribute__((ext_vector_type(8)));
typedef int      v4i  __attribute__((ext_vector_type(4)));

#ifdef ASYNC_LDS
// builtin signature (from hipcc diagnostic): (global v4i*, local v4i*, imm offset, imm cpol)
typedef __attribute__((address_space(1))) v4i* gptr128;
typedef __attribute__((address_space(3))) v4i* lptr128;
#endif

// A-fragment (16x32 f16): lane L holds row M=L%16; halves h -> K = h + (L>=16?16:0)
// => two contiguous 8-half (16B) chunks at k0+koff and k0+koff+8.
__device__ __forceinline__ v16h ldA(const _Float16* p, int ld, int m0, int k0, int lane) {
  const _Float16* row = p + (m0 + (lane & 15)) * ld + k0 + ((lane & 16) ? 16 : 0);
  union { v16h v; v8h h[2]; } u;
  u.h[0] = *(const v8h*)(row);
  u.h[1] = *(const v8h*)(row + 8);
  return u.v;
}
// B-fragment (32x16 f16): lane L holds column n=L%16, rows k0+(L>=16?16:0)..+15 contiguous
// when the source is stored n-major ([n][k], "transposed weight") => one 32B load.
__device__ __forceinline__ v16h ldB(const _Float16* p, int ld, int n0, int k0, int lane) {
  return *(const v16h*)(p + (n0 + (lane & 15)) * ld + k0 + ((lane & 16) ? 16 : 0));
}

#define WMMA(a,b,c) __builtin_amdgcn_wmma_f32_16x16x32_f16(false,(a),false,(b),(short)0,(c),false,false)

__device__ __forceinline__ float geluf(float v) {
  return 0.5f * v * (1.0f + erff(v * 0.70710678118654752f));
}

// LayerNorm over 64 rows of tokf -> f16 out. 4 threads per row, 64 cols each.
__device__ void ln_rows(const float (*in)[DD], _Float16 (*out)[DD],
                        const float* __restrict__ g, const float* __restrict__ bta,
                        float (*red)[4], float (*red2)[4], int tid) {
  int r = tid >> 2, qn = tid & 3, c0 = qn * 64;
  float s = 0.f, s2 = 0.f;
  for (int c = 0; c < 64; ++c) { float v = in[r][c0 + c]; s += v; s2 += v * v; }
  red[r][qn] = s; red2[r][qn] = s2;
  __syncthreads();
  float mean = (red[r][0] + red[r][1] + red[r][2] + red[r][3]) * (1.0f / DD);
  float var  = (red2[r][0] + red2[r][1] + red2[r][2] + red2[r][3]) * (1.0f / DD) - mean * mean;
  float inv = rsqrtf(var + 1e-5f);
  for (int c = 0; c < 64; ++c) {
    int cc = c0 + c;
    out[r][cc] = (_Float16)(((in[r][cc] - mean) * inv) * g[cc] + bta[cc]);
  }
  __syncthreads();
}

// Fused Q/K/V projection for one head: each wave owns one nj column-tile and an mi pair;
// A fragments are reused across 3 WMMAs, B fragments across 2.
__device__ void gemm_qkv_head(const _Float16 (*A)[DD],
                              const _Float16* __restrict__ wqt,
                              const _Float16* __restrict__ wkt,
                              const _Float16* __restrict__ wvt,
                              const float* __restrict__ bq,
                              const float* __restrict__ bk,
                              const float* __restrict__ bv,
                              int nbase,
                              _Float16 (*qh)[DHD], _Float16 (*kh)[DHD], _Float16 (*vth)[DHD],
                              int wave, int lane) {
  const int nj  = wave & 3;
  const int mi0 = (wave >> 2) * 2;
  v8f aq[2] = {}, ak[2] = {}, av[2] = {};
#pragma unroll
  for (int k0 = 0; k0 < DD; k0 += 32) {
    v16h bqf = ldB(wqt, DD, nbase + nj * 16, k0, lane);
    v16h bkf = ldB(wkt, DD, nbase + nj * 16, k0, lane);
    v16h bvf = ldB(wvt, DD, nbase + nj * 16, k0, lane);
#pragma unroll
    for (int m = 0; m < 2; ++m) {
      v16h a = ldA(&A[0][0], DD, (mi0 + m) * 16, k0, lane);
      aq[m] = WMMA(a, bqf, aq[m]);
      ak[m] = WMMA(a, bkf, ak[m]);
      av[m] = WMMA(a, bvf, av[m]);
    }
  }
  const int n = nj * 16 + (lane & 15);
  const float bnq = bq[nbase + n], bnk = bk[nbase + n], bnv = bv[nbase + n];
#pragma unroll
  for (int m = 0; m < 2; ++m) {
    int mb = (mi0 + m) * 16 + ((lane & 16) ? 8 : 0);
#pragma unroll
    for (int j = 0; j < 8; ++j) {
      qh[mb + j][n]  = (_Float16)(aq[m][j] + bnq);
      kh[mb + j][n]  = (_Float16)(ak[m][j] + bnk);
      vth[n][mb + j] = (_Float16)(av[m][j] + bnv);   // V stored transposed [dh][tok]
    }
  }
}

extern "C" __global__ void __launch_bounds__(256)
matphys_fused(const float* __restrict__ x, const float* __restrict__ elem,
              const float* __restrict__ fw, const float* __restrict__ fb,
              const float* __restrict__ alphap,
              const float* __restrict__ ln1g, const float* __restrict__ ln1b,
              const float* __restrict__ bq, const float* __restrict__ bk,
              const float* __restrict__ bvv, const float* __restrict__ bo,
              const float* __restrict__ ln2g, const float* __restrict__ ln2b,
              const float* __restrict__ b1, const float* __restrict__ b2,
              const float* __restrict__ se1w, const float* __restrict__ se1b,
              const float* __restrict__ se2w, const float* __restrict__ se2b,
              const float* __restrict__ ln3g, const float* __restrict__ ln3b,
              const float* __restrict__ h1w, const float* __restrict__ h1b,
              const float* __restrict__ h2w, const float* __restrict__ h2b,
              const _Float16* __restrict__ wqt, const _Float16* __restrict__ wkt,
              const _Float16* __restrict__ wvt, const _Float16* __restrict__ wot,
              const _Float16* __restrict__ w1t, const _Float16* __restrict__ w2t,
              const float* __restrict__ biasg, float* __restrict__ outp) {
  __shared__ float    tokf[NP][DD];                                 // 64KB residual (f32)
  __shared__ _Float16 hbuf[NP][DD];                                 // 32KB LN output (f16)
  __shared__ __align__(32) unsigned char attws[NP * DHD * 2 * 3 + NP * NP * 4]; // 40KB
  __shared__ __align__(16) float biasl[NP][NP];                     // 16KB bias tile in LDS
  __shared__ float red[NP][4];
  __shared__ float red2[NP][4];
  __shared__ float xrow[NP], mfr[NP];
  __shared__ float zbuf[DD], sgate[DD], tse[64], hh3[DD], g1b[128];
  __shared__ float stat[2];

  _Float16 (*qh)[DHD]    = (_Float16(*)[DHD])(attws);           // Q / later y_h
  _Float16 (*kh)[DHD]    = (_Float16(*)[DHD])(attws + 8192);    // K / later softmax probs
  _Float16 (*vth)[DHD]   = (_Float16(*)[DHD])(attws + 16384);   // V^T [dh][tok]
  float    (*scores)[NP] = (float(*)[NP])(attws + 24576);       // f32 scores
  _Float16 (*gbuf)[DD]   = (_Float16(*)[DD])(attws);            // FFN chunk (aliases attn bufs)

  const int tid  = threadIdx.x;
  const int lane = tid & 31;
  const int wave = tid >> 5;
  const int b    = blockIdx.x;

#ifdef ASYNC_LDS
  {
    // stage bias[64][64] f32 into LDS with the CDNA5 async global->LDS path:
    // 256 threads x 4 x b128 = 16KB, tracked by ASYNCcnt.
    const float* gsrc = biasg + tid * 16;
    float*       ldst = &biasl[0][0] + tid * 16;
    __builtin_amdgcn_global_load_async_to_lds_b128((gptr128)(gsrc + 0),  (lptr128)(ldst + 0),  0, 0);
    __builtin_amdgcn_global_load_async_to_lds_b128((gptr128)(gsrc + 4),  (lptr128)(ldst + 4),  0, 0);
    __builtin_amdgcn_global_load_async_to_lds_b128((gptr128)(gsrc + 8),  (lptr128)(ldst + 8),  0, 0);
    __builtin_amdgcn_global_load_async_to_lds_b128((gptr128)(gsrc + 12), (lptr128)(ldst + 12), 0, 0);
  }
#endif

  if (tid < NP) {
    float xv = (tid < NTOK) ? x[b * NTOK + tid] : 0.0f;
    xrow[tid] = xv;
    mfr[tid]  = (xv > 1e-6f) ? 1.0f : 0.0f;
  }
  __syncthreads();
  if (tid == 0) {
    float s = 0.f;
    for (int n = 0; n < NP; ++n) s += mfr[n];
    stat[0] = fmaxf(s, 1e-8f);   // masked count
  }
  // --- token embedding: elem + x*frac_w + frac_b (rows >= NTOK zeroed) ---
  for (int n = 0; n < NP; ++n) {
    float v = 0.0f;
    if (n < NTOK) v = elem[n * DD + tid] + xrow[n] * fw[tid] + fb[tid];
    tokf[n][tid] = v;
  }
#ifndef ASYNC_LDS
  for (int i = 0; i < 16; ++i) ((float*)biasl)[tid + i * 256] = biasg[tid + i * 256];
#endif
  __syncthreads();

  ln_rows(tokf, hbuf, ln1g, ln1b, red, red2, tid);

#ifdef ASYNC_LDS
#  ifdef HAVE_WAIT_ASYNC
  __builtin_amdgcn_s_wait_asynccnt(0);
#  else
  asm volatile("s_wait_asynccnt 0" ::: "memory");
#  endif
  __syncthreads();
#endif

  const float alpha = alphap[0];

  // ================= attention, head by head =================
  for (int hd = 0; hd < HH; ++hd) {
    const int nb = hd * DHD;
    gemm_qkv_head(hbuf, wqt, wkt, wvt, bq, bk, bvv, nb, qh, kh, vth, wave, lane);
    __syncthreads();
    // scores = Q Kt / 8 + alpha*bias (K rows used directly as B operand)
    {
      const int nj = wave & 3, mi0 = (wave >> 2) * 2;
      v8f s[2] = {};
#pragma unroll
      for (int k0 = 0; k0 < DHD; k0 += 32) {
        v16h bb = ldB(&kh[0][0], DHD, nj * 16, k0, lane);
#pragma unroll
        for (int m = 0; m < 2; ++m) {
          v16h a = ldA(&qh[0][0], DHD, (mi0 + m) * 16, k0, lane);
          s[m] = WMMA(a, bb, s[m]);
        }
      }
      int n = nj * 16 + (lane & 15);
#pragma unroll
      for (int m = 0; m < 2; ++m) {
        int mb = (mi0 + m) * 16 + ((lane & 16) ? 8 : 0);
#pragma unroll
        for (int j = 0; j < 8; ++j) {
          int mm = mb + j;
          scores[mm][n] = s[m][j] * 0.125f + alpha * biasl[mm][n];
        }
      }
    }
    __syncthreads();
    // masked softmax -> probs f16 into kh
    {
      int r = tid >> 2, c0 = (tid & 3) * 16;
      float vreg[16];
      float lmax = -3.0e38f;
      for (int c = 0; c < 16; ++c) {
        float v = (mfr[c0 + c] > 0.5f) ? scores[r][c0 + c] : -3.0e38f;
        vreg[c] = v; lmax = fmaxf(lmax, v);
      }
      red[r][tid & 3] = lmax;
      __syncthreads();
      float rmax = fmaxf(fmaxf(red[r][0], red[r][1]), fmaxf(red[r][2], red[r][3]));
      float lsum = 0.f;
      for (int c = 0; c < 16; ++c) { vreg[c] = expf(vreg[c] - rmax); lsum += vreg[c]; }
      red2[r][tid & 3] = lsum;
      __syncthreads();
      float inv = 1.0f / fmaxf(red2[r][0] + red2[r][1] + red2[r][2] + red2[r][3], 1e-20f);
      for (int c = 0; c < 16; ++c) kh[r][c0 + c] = (_Float16)(vreg[c] * inv);
    }
    __syncthreads();
    // y_h = probs @ V  -> qh (plain [tok][dh])
    {
      const int nj = wave & 3, mi0 = (wave >> 2) * 2;
      v8f s[2] = {};
#pragma unroll
      for (int k0 = 0; k0 < NP; k0 += 32) {
        v16h bb = ldB(&vth[0][0], NP, nj * 16, k0, lane);
#pragma unroll
        for (int m = 0; m < 2; ++m) {
          v16h a = ldA(&kh[0][0], DHD, (mi0 + m) * 16, k0, lane);
          s[m] = WMMA(a, bb, s[m]);
        }
      }
      int n = nj * 16 + (lane & 15);
#pragma unroll
      for (int m = 0; m < 2; ++m) {
        int mb = (mi0 + m) * 16 + ((lane & 16) ? 8 : 0);
#pragma unroll
        for (int j = 0; j < 8; ++j) qh[mb + j][n] = (_Float16)s[m][j];
      }
    }
    __syncthreads();
    // tokf += y_h @ Wo[hd*64 .. , :]; wave owns nj pair, B frags reused across 4 mi tiles
    {
      const int nj0 = wave * 2;
      v8f acc[8] = {};
#pragma unroll
      for (int k0 = 0; k0 < DHD; k0 += 32) {
        v16h b0 = ldB(wot, DD, nj0 * 16, nb + k0, lane);
        v16h b1 = ldB(wot, DD, (nj0 + 1) * 16, nb + k0, lane);
#pragma unroll
        for (int mi = 0; mi < 4; ++mi) {
          v16h a = ldA(&qh[0][0], DHD, mi * 16, k0, lane);
          acc[mi * 2]     = WMMA(a, b0, acc[mi * 2]);
          acc[mi * 2 + 1] = WMMA(a, b1, acc[mi * 2 + 1]);
        }
      }
#pragma unroll
      for (int mi = 0; mi < 4; ++mi)
#pragma unroll
        for (int u = 0; u < 2; ++u) {
          int n  = (nj0 + u) * 16 + (lane & 15);
          int mb = mi * 16 + ((lane & 16) ? 8 : 0);
          float badd = (hd == 0) ? bo[n] : 0.0f;
#pragma unroll
          for (int j = 0; j < 8; ++j) tokf[mb + j][n] += acc[mi * 2 + u][j] + badd;
        }
    }
    __syncthreads();
  }

  // ================= FFN, 4 chunks of 256 DFF columns =================
  ln_rows(tokf, hbuf, ln2g, ln2b, red, red2, tid);
  for (int ch = 0; ch < 4; ++ch) {
    {   // G = gelu(h @ W1[:,chunk] + b1)
      const int nj0 = wave * 2;
      v8f acc[8] = {};
#pragma unroll
      for (int k0 = 0; k0 < DD; k0 += 32) {
        v16h b0 = ldB(w1t, DD, ch * 256 + nj0 * 16, k0, lane);
        v16h b1 = ldB(w1t, DD, ch * 256 + (nj0 + 1) * 16, k0, lane);
#pragma unroll
        for (int mi = 0; mi < 4; ++mi) {
          v16h a = ldA(&hbuf[0][0], DD, mi * 16, k0, lane);
          acc[mi * 2]     = WMMA(a, b0, acc[mi * 2]);
          acc[mi * 2 + 1] = WMMA(a, b1, acc[mi * 2 + 1]);
        }
      }
#pragma unroll
      for (int mi = 0; mi < 4; ++mi)
#pragma unroll
        for (int u = 0; u < 2; ++u) {
          int n  = (nj0 + u) * 16 + (lane & 15);
          int mb = mi * 16 + ((lane & 16) ? 8 : 0);
          float bn = b1[ch * 256 + n];
#pragma unroll
          for (int j = 0; j < 8; ++j) gbuf[mb + j][n] = (_Float16)geluf(acc[mi * 2 + u][j] + bn);
        }
    }
    __syncthreads();
    {   // tokf += G @ W2[chunk,:] (+ b2 once)
      const int nj0 = wave * 2;
      v8f acc[8] = {};
#pragma unroll
      for (int k0 = 0; k0 < DD; k0 += 32) {
        v16h b0 = ldB(w2t, DFF, nj0 * 16, ch * 256 + k0, lane);
        v16h b1 = ldB(w2t, DFF, (nj0 + 1) * 16, ch * 256 + k0, lane);
#pragma unroll
        for (int mi = 0; mi < 4; ++mi) {
          v16h a = ldA(&gbuf[0][0], DD, mi * 16, k0, lane);
          acc[mi * 2]     = WMMA(a, b0, acc[mi * 2]);
          acc[mi * 2 + 1] = WMMA(a, b1, acc[mi * 2 + 1]);
        }
      }
#pragma unroll
      for (int mi = 0; mi < 4; ++mi)
#pragma unroll
        for (int u = 0; u < 2; ++u) {
          int n  = (nj0 + u) * 16 + (lane & 15);
          int mb = mi * 16 + ((lane & 16) ? 8 : 0);
          float bn = (ch == 0) ? b2[n] : 0.0f;
#pragma unroll
          for (int j = 0; j < 8; ++j) tokf[mb + j][n] += acc[mi * 2 + u][j] + bn;
        }
    }
    __syncthreads();
  }

  // ================= SE + pooling + head =================
  {
    float acc = 0.f;
    for (int n = 0; n < NTOK; ++n) acc += tokf[n][tid] * mfr[n];
    zbuf[tid] = acc / stat[0];     // masked mean z
  }
  __syncthreads();
  if (tid < 64) {
    float acc = 0.f;
    for (int k = 0; k < DD; ++k) acc += zbuf[k] * se1w[k * 64 + tid];
    tse[tid] = geluf(acc + se1b[tid]);
  }
  __syncthreads();
  {
    float acc = 0.f;
    for (int k = 0; k < 64; ++k) acc += tse[k] * se2w[k * DD + tid];
    sgate[tid] = 1.0f / (1.0f + expf(-(acc + se2b[tid])));
  }
  __syncthreads();
  // pooled = masked_mean(tok * s) = s * z ; then LN3
  hh3[tid] = sgate[tid] * zbuf[tid];
  __syncthreads();
  if (tid < 64) {
    float s = 0.f, s2 = 0.f;
    for (int c = 0; c < 4; ++c) { float v = hh3[tid * 4 + c]; s += v; s2 += v * v; }
    red[tid][0] = s; red2[tid][0] = s2;
  }
  __syncthreads();
  if (tid == 0) {
    float s = 0.f, s2 = 0.f;
    for (int r = 0; r < 64; ++r) { s += red[r][0]; s2 += red2[r][0]; }
    float mean = s * (1.0f / DD);
    float var  = s2 * (1.0f / DD) - mean * mean;
    stat[0] = mean; stat[1] = rsqrtf(var + 1e-5f);
  }
  __syncthreads();
  hh3[tid] = (hh3[tid] - stat[0]) * stat[1] * ln3g[tid] + ln3b[tid];
  __syncthreads();
  if (tid < 128) {
    float acc = 0.f;
    for (int k = 0; k < DD; ++k) acc += hh3[k] * h1w[k * 128 + tid];
    g1b[tid] = geluf(acc + h1b[tid]);
  }
  __syncthreads();
  if (tid == 0) {
    float acc = 0.f;
    for (int k = 0; k < 128; ++k) acc += g1b[k] * h2w[k];
    float v = acc + h2b[0];
    outp[b] = (v > 20.0f) ? v : ((v < -20.0f) ? expf(v) : log1pf(expf(v)));
  }
}

// ---- weight transpose+convert: src f32 [K][Nn] -> dst f16 [Nn][K] ----
extern "C" __global__ void conv_t(const float* __restrict__ src, _Float16* __restrict__ dst,
                                  int K, int Nn) {
  int i = blockIdx.x * 256 + threadIdx.x;
  if (i < K * Nn) {
    int k = i / Nn, n = i - k * Nn;
    dst[n * K + k] = (_Float16)src[i];
  }
}

// ---- pairwise primitive-distance bias: out[64][64] f32 (padded rows/cols = 0) ----
extern "C" __global__ void bias_k(const float* __restrict__ P, const float* __restrict__ bw,
                                  const float* __restrict__ bb, float* __restrict__ outb) {
  int i = blockIdx.x, j = threadIdx.x;
  float v = 0.f;
  if (i < NTOK && j < NTOK) {
    float acc = 0.f;
    for (int k = 0; k < 8; ++k) { float d = P[i * 8 + k] - P[j * 8 + k]; acc += d * d * bw[k]; }
    v = acc + bb[0];
  }
  outb[i * NP + j] = v;
}

extern "C" void kernel_launch(void* const* d_in, const int* in_sizes, int n_in,
                              void* d_out, int out_size, void* d_ws, size_t ws_size,
                              hipStream_t stream) {
  (void)in_sizes; (void)n_in; (void)out_size; (void)ws_size;
  const float* x      = (const float*)d_in[0];
  const float* elem   = (const float*)d_in[1];
  const float* fw     = (const float*)d_in[2];
  const float* fb     = (const float*)d_in[3];
  const float* P      = (const float*)d_in[4];
  const float* bw     = (const float*)d_in[5];
  const float* bb     = (const float*)d_in[6];
  const float* alphap = (const float*)d_in[7];
  const float* ln1g   = (const float*)d_in[8];
  const float* ln1b   = (const float*)d_in[9];
  const float* Wq     = (const float*)d_in[10];
  const float* bq     = (const float*)d_in[11];
  const float* Wk     = (const float*)d_in[12];
  const float* bk     = (const float*)d_in[13];
  const float* Wv     = (const float*)d_in[14];
  const float* bv     = (const float*)d_in[15];
  const float* Wo     = (const float*)d_in[16];
  const float* bo     = (const float*)d_in[17];
  const float* ln2g   = (const float*)d_in[18];
  const float* ln2b   = (const float*)d_in[19];
  const float* W1     = (const float*)d_in[20];
  const float* b1     = (const float*)d_in[21];
  const float* W2     = (const float*)d_in[22];
  const float* b2     = (const float*)d_in[23];
  const float* se1w   = (const float*)d_in[24];
  const float* se1b   = (const float*)d_in[25];
  const float* se2w   = (const float*)d_in[26];
  const float* se2b   = (const float*)d_in[27];
  const float* ln3g   = (const float*)d_in[28];
  const float* ln3b   = (const float*)d_in[29];
  const float* h1w    = (const float*)d_in[30];
  const float* h1b    = (const float*)d_in[31];
  const float* h2w    = (const float*)d_in[32];
  const float* h2b    = (const float*)d_in[33];

  _Float16* wqt = (_Float16*)d_ws;
  _Float16* wkt = wqt + 256 * 256;
  _Float16* wvt = wkt + 256 * 256;
  _Float16* wot = wvt + 256 * 256;
  _Float16* w1t = wot + 256 * 256;         // [1024][256]
  _Float16* w2t = w1t + 256 * 1024;        // [256][1024]
  float*   biasg = (float*)(w2t + 1024 * 256);  // [64][64], offset 1.5MB (256B aligned)

  conv_t<<<256, 256, 0, stream>>>(Wq, wqt, 256, 256);
  conv_t<<<256, 256, 0, stream>>>(Wk, wkt, 256, 256);
  conv_t<<<256, 256, 0, stream>>>(Wv, wvt, 256, 256);
  conv_t<<<256, 256, 0, stream>>>(Wo, wot, 256, 256);
  conv_t<<<1024, 256, 0, stream>>>(W1, w1t, 256, 1024);
  conv_t<<<1024, 256, 0, stream>>>(W2, w2t, 1024, 256);
  bias_k<<<64, 64, 0, stream>>>(P, bw, bb, biasg);

  matphys_fused<<<4096, 256, 0, stream>>>(x, elem, fw, fb, alphap, ln1g, ln1b,
      bq, bk, bv, bo, ln2g, ln2b, b1, b2, se1w, se1b, se2w, se2b, ln3g, ln3b,
      h1w, h1b, h2w, h2b, wqt, wkt, wvt, wot, w1t, w2t, biasg, (float*)d_out);
}
